// ReluFinefy_28664611733900
// MI455X (gfx1250) — matrix operand order repaired and checked by reading, not compiled, not run on hardware
//
#include <hip/hip_runtime.h>

// ---------------------------------------------------------------------------
// ReLU + Finefy (gather + [Nf,288]x[288,32] GEMM) for gfx1250 (MI455X)
//
// K-block (32) == C_IN: each WMMA A-fragment is two contiguous 16B chunks of
// one gathered bf16 coarse row -> gather L2 -> VGPR -> WMMA directly.
// Weights pre-converted/transposed/padded to bf16 once; staged per block in
// LDS with a b128 memcpy; each B fragment feeds 4 M-tiles (64 rows/wave,
// 256/block). Bias is pre-seeded into the WMMA accumulators.
// ---------------------------------------------------------------------------

typedef __attribute__((ext_vector_type(16))) __bf16         v16bf;
typedef __attribute__((ext_vector_type(8)))  float          v8f;
typedef __attribute__((ext_vector_type(4)))  int            i4;
typedef __attribute__((ext_vector_type(4)))  float          f4;

union BFrag { i4 q[2]; v16bf v; };

__device__ __forceinline__ unsigned short f2bf_rne(float x) {
    unsigned int u = __float_as_uint(x);
    u += 0x7FFFu + ((u >> 16) & 1u);          // round-to-nearest-even
    return (unsigned short)(u >> 16);
}

// relu + convert 8 floats -> one 16B half of an A fragment (f32 fallback only)
__device__ __forceinline__ i4 pack_relu_bf8(f4 a, f4 b) {
    union { unsigned short u[8]; i4 q; } r;
    r.u[0] = f2bf_rne(fmaxf(a.x, 0.0f));
    r.u[1] = f2bf_rne(fmaxf(a.y, 0.0f));
    r.u[2] = f2bf_rne(fmaxf(a.z, 0.0f));
    r.u[3] = f2bf_rne(fmaxf(a.w, 0.0f));
    r.u[4] = f2bf_rne(fmaxf(b.x, 0.0f));
    r.u[5] = f2bf_rne(fmaxf(b.y, 0.0f));
    r.u[6] = f2bf_rne(fmaxf(b.z, 0.0f));
    r.u[7] = f2bf_rne(fmaxf(b.w, 0.0f));
    return r.q;
}

#define KNB   9                  // neighbors per fine vertex (= K blocks)
#define CIN   32                 // input channels (= WMMA K per block)
#define NOUT  32                 // output filters
#define KTOT  (KNB * CIN)        // 288
#define WPAD  296                // padded K stride for sW -> conflict-free b128
#define WAVES 4
#define TILES 4                  // M tiles per wave
#define RPW   (TILES * 16)       // 64 rows per wave
#define RPB   (WAVES * RPW)      // 256 rows per block

// Pre-pass A: dst = bf16(relu(src)), 4 elements/thread.
__global__ void relu_cvt_bf16_kernel(const float* __restrict__ src,
                                     unsigned short* __restrict__ dst, int n4)
{
    int i = blockIdx.x * blockDim.x + threadIdx.x;
    if (i < n4) {
        f4 v = ((const f4*)src)[i];
        union { unsigned short u[4]; long long q; } h;
        h.u[0] = f2bf_rne(fmaxf(v.x, 0.0f));
        h.u[1] = f2bf_rne(fmaxf(v.y, 0.0f));
        h.u[2] = f2bf_rne(fmaxf(v.z, 0.0f));
        h.u[3] = f2bf_rne(fmaxf(v.w, 0.0f));
        ((long long*)dst)[i] = h.q;
    }
}

// Pre-pass B: wt[n][k] = bf16(wgt[k][n]), transposed + padded to WPAD.
__global__ void cvt_wgt_t_kernel(const float* __restrict__ wgt,
                                 unsigned short* __restrict__ wt)
{
    int j = blockIdx.x * blockDim.x + threadIdx.x;   // 0 .. 9215
    if (j < KTOT * NOUT) {
        int k = j >> 5;
        int n = j & (NOUT - 1);
        wt[n * WPAD + k] = f2bf_rne(wgt[j]);
    }
}

template <bool USE_BF>
__global__ __launch_bounds__(WAVES * 32)
void finefy_wmma_kernel(const float* __restrict__ lvc_f32,
                        const unsigned short* __restrict__ lvc_bf,
                        const int* __restrict__ nidx,
                        const float* __restrict__ wgt,
                        const unsigned short* __restrict__ wgt_t,
                        const float* __restrict__ bias,
                        float* __restrict__ out,
                        int nf)
{
    __shared__ alignas(16) unsigned short sW[NOUT][WPAD];          // 18,944 B
    __shared__ alignas(16) int sIdx[WAVES][KNB][16][TILES];        //  9,216 B
    __shared__ float sBias[NOUT];

    const int  tid      = threadIdx.x;
    const int  wave     = tid >> 5;
    const int  lane     = tid & 31;
    const int  waveRow0 = blockIdx.x * RPB + wave * RPW;
    const bool fullWave = (waveRow0 + RPW) <= nf;

    // ---- stage weights ----
    if (USE_BF) {
        // pre-transposed/padded in scratch: straight b128 memcpy
        i4*       dst = (i4*)&sW[0][0];
        const i4* src = (const i4*)wgt_t;
        for (int j = tid; j < (NOUT * WPAD) / 8; j += WAVES * 32)
            dst[j] = src[j];
    } else {
        for (int j = tid; j < KTOT * NOUT; j += WAVES * 32)
            sW[j & (NOUT - 1)][j >> 5] = f2bf_rne(wgt[j]);
    }
    if (tid < NOUT) sBias[tid] = bias[tid];

    // ---- stage this wave's 576 neighbor indices, transposed [kb][m][tile] ----
    {
        const unsigned base = (unsigned)waveRow0 * KNB;
        if (fullWave) {
            for (int t = lane; t < RPW * KNB; t += 32) {
                int r = t / KNB, kb = t - r * KNB;
                sIdx[wave][kb][r & 15][r >> 4] = nidx[base + t];
            }
        } else {
            for (int t = lane; t < RPW * KNB; t += 32) {
                int r = t / KNB, kb = t - r * KNB;
                sIdx[wave][kb][r & 15][r >> 4] =
                    ((waveRow0 + r) < nf) ? nidx[base + t] : 0;
            }
        }
    }
    __syncthreads();

    const int half = lane >> 4;               // 0: lanes 0-15, 1: lanes 16-31
    const int mrow = lane & 15;

    // bias pre-seeded into accumulators (acc column N is constant per lane)
    const float bv0 = sBias[mrow];
    const float bv1 = sBias[mrow + 16];
    const v8f   s0  = {bv0, bv0, bv0, bv0, bv0, bv0, bv0, bv0};
    const v8f   s1  = {bv1, bv1, bv1, bv1, bv1, bv1, bv1, bv1};
    v8f acc[TILES][2];
    #pragma unroll
    for (int tt = 0; tt < TILES; ++tt) { acc[tt][0] = s0; acc[tt][1] = s1; }

    const unsigned short* bRow0 = sW[mrow];        // column N = mrow
    const unsigned short* bRow1 = sW[mrow + 16];   // column N = mrow + 16
    const unsigned char*  abase = USE_BF ? (const unsigned char*)lvc_bf
                                         : (const unsigned char*)lvc_f32;

    #pragma unroll
    for (int kb = 0; kb < KNB; ++kb) {
        // B fragments (32x16 bf16): lane<16 holds K {0..7,16..23}, lane>=16 {8..15,24..31}
        const int bOff = kb * 32 + half * 8;
        BFrag b0, b1;
        b0.q[0] = *(const i4*)(bRow0 + bOff);
        b0.q[1] = *(const i4*)(bRow0 + bOff + 16);
        b1.q[0] = *(const i4*)(bRow1 + bOff);
        b1.q[1] = *(const i4*)(bRow1 + bOff + 16);

        // 4 tiles' coarse-row indices for (kb, mrow) in one b128 LDS read
        const i4 cr4 = *(const i4*)&sIdx[wave][kb][mrow][0];

        #pragma unroll
        for (int tt = 0; tt < TILES; ++tt) {
            BFrag a;   // A (16x32): lane<16 ch {0..7,16..23}, lane>=16 {8..15,24..31}
            if (USE_BF) {
                const unsigned off = (unsigned)cr4[tt] * (CIN * 2u)
                                   + (unsigned)half * 16u;      // byte offset, <16MB
                a.q[0] = *(const i4*)(abase + off);
                a.q[1] = *(const i4*)(abase + off + 32);
            } else {
                const unsigned off = (unsigned)cr4[tt] * (CIN * 4u)
                                   + (unsigned)half * 32u;
                const f4* fp = (const f4*)(abase + off);
                a.q[0] = pack_relu_bf8(fp[0], fp[1]);
                a.q[1] = pack_relu_bf8(fp[4], fp[5]);
            }
            acc[tt][0] = __builtin_amdgcn_wmma_f32_16x16x32_bf16(
                false, a.v, false, b0.v, (short)0, acc[tt][0], false, false);
            acc[tt][1] = __builtin_amdgcn_wmma_f32_16x16x32_bf16(
                false, a.v, false, b1.v, (short)0, acc[tt][1], false, false);
        }
    }

    // ---- store (C layout: VGPR r -> M = r + 8*half, N = mrow) ----
    const unsigned obase = (unsigned)(waveRow0 + half * 8) * NOUT + (unsigned)mrow;
    if (fullWave) {
        #pragma unroll
        for (int tt = 0; tt < TILES; ++tt) {
            #pragma unroll
            for (int r = 0; r < 8; ++r) {
                float* o = out + obase + (unsigned)(tt * 16 + r) * NOUT;
                o[0]  = acc[tt][0][r];
                o[16] = acc[tt][1][r];
            }
        }
    } else {
        #pragma unroll
        for (int tt = 0; tt < TILES; ++tt) {
            #pragma unroll
            for (int r = 0; r < 8; ++r) {
                int fineRow = waveRow0 + tt * 16 + half * 8 + r;
                if (fineRow < nf) {
                    float* o = out + (unsigned)fineRow * NOUT + (unsigned)mrow;
                    o[0]  = acc[tt][0][r];
                    o[16] = acc[tt][1][r];
                }
            }
        }
    }
}

extern "C" void kernel_launch(void* const* d_in, const int* in_sizes, int n_in,
                              void* d_out, int out_size, void* d_ws, size_t ws_size,
                              hipStream_t stream)
{
    const float* lvc  = (const float*)d_in[0];   // [Nc, 32] f32
    const int*   nidx = (const int*)  d_in[1];   // [Nf, 9]  i32
    const float* wgt  = (const float*)d_in[2];   // [288, 32] f32
    const float* bias = (const float*)d_in[3];   // [32] f32
    float* out = (float*)d_out;                  // [Nf, 32] f32

    const int    ncElems = in_sizes[0];          // Nc * 32
    const int    nf      = in_sizes[1] / KNB;    // Nf
    const size_t lvBytes = (size_t)ncElems * sizeof(unsigned short);
    const size_t wBytes  = (size_t)NOUT * WPAD * sizeof(unsigned short);
    const int    grid    = (nf + RPB - 1) / RPB;

    if (ws_size >= lvBytes + wBytes) {
        unsigned short* lvbf = (unsigned short*)d_ws;
        unsigned short* wt   = lvbf + ncElems;
        const int n4 = ncElems / 4;
        relu_cvt_bf16_kernel<<<(n4 + 255) / 256, 256, 0, stream>>>(lvc, lvbf, n4);
        cvt_wgt_t_kernel<<<(KTOT * NOUT + 255) / 256, 256, 0, stream>>>(wgt, wt);
        finefy_wmma_kernel<true><<<grid, WAVES * 32, 0, stream>>>(
            lvc, lvbf, nidx, wgt, wt, bias, out, nf);
    } else {
        finefy_wmma_kernel<false><<<grid, WAVES * 32, 0, stream>>>(
            lvc, nullptr, nidx, wgt, nullptr, bias, out, nf);
    }
}
